// WritePath_1597727834710
// MI455X (gfx1250) — compile-verified, matrix operand order checked by reading, not run on hardware
//
#include <hip/hip_runtime.h>
#include <math.h>

// ---------------- problem constants ----------------
constexpr int Bb = 4;
constexpr int Tt = 4096;
constexpr int Hh = 2048;
constexpr int Dd = 256;
constexpr int Nn = 32768;
constexpr int HQ = 512;   // Hq = H/4

// ---------------- WMMA types ----------------
typedef __attribute__((ext_vector_type(16))) __bf16 bf16x16;
typedef __attribute__((ext_vector_type(8)))  float  floatx8;

union FragU { unsigned int u[8]; bf16x16 v; };

__device__ __forceinline__ unsigned short f2bf(float x) {
    unsigned int u = __float_as_uint(x);
    u += 0x7FFFu + ((u >> 16) & 1u);      // round-to-nearest-even
    return (unsigned short)(u >> 16);
}
__device__ __forceinline__ unsigned int pack2bf(float lo, float hi) {
    return (unsigned int)f2bf(lo) | ((unsigned int)f2bf(hi) << 16);
}

// A/B fragment gather for v_wmma_*_16x16x32_bf16 from a row-major [16 x K] tile.
// Lane L holds row m = L&15; half-wave selects K base 0 / 8; VGPR v holds the
// K pair (v>=4?16:0) + (v&3)*2 (+8 for upper half-wave).  (ISA 7.12.2 tables)
__device__ __forceinline__ bf16x16 frag_bf16(const unsigned short* __restrict__ base,
                                             int ld, int lane) {
    const unsigned short* row = base + (size_t)(lane & 15) * ld + ((lane >> 4) & 1) * 8;
    FragU f;
#pragma unroll
    for (int v = 0; v < 8; ++v) {
        int k = ((v & 4) << 2) + ((v & 3) << 1);
        f.u[v] = *(const unsigned int*)(row + k);
    }
    return f.v;
}
// Fallback gather converting f32 source on the fly (only used if ws is too
// small to hold a pre-converted bf16 copy of `hidden`).
__device__ __forceinline__ bf16x16 frag_f32(const float* __restrict__ base,
                                            int ld, int lane) {
    const float* row = base + (size_t)(lane & 15) * ld + ((lane >> 4) & 1) * 8;
    FragU f;
#pragma unroll
    for (int v = 0; v < 8; ++v) {
        int k = ((v & 4) << 2) + ((v & 3) << 1);
        f.u[v] = pack2bf(row[k], row[k + 1]);
    }
    return f.v;
}

__device__ __forceinline__ floatx8 wmma_bf16(bf16x16 a, bf16x16 b, floatx8 c) {
    return __builtin_amdgcn_wmma_f32_16x16x32_bf16(false, a, false, b,
                                                   (short)0, c, false, false);
}

// ---------------- kernels ----------------

__global__ void k_init(float* prec_sum, unsigned long long* bestpack, int* anyflag) {
    int i = blockIdx.x * blockDim.x + threadIdx.x;
    if (i < Tt) { prec_sum[i] = 0.0f; bestpack[i] = 0ull; }
    if (i == 0) *anyflag = 0;
}

__global__ void k_any(const int* __restrict__ mask, int* anyflag) {
    int i = blockIdx.x * blockDim.x + threadIdx.x;
    int v = (i < Nn) ? mask[i] : 0;
    unsigned long long bal = __ballot(v != 0);
    if ((threadIdx.x & 31) == 0 && bal != 0ull) atomicOr(anyflag, 1);
}

// Weights f32 -> bf16, two elements per 32-bit store.
__global__ void k_cvtw(const float* __restrict__ wobs, const float* __restrict__ wg1,
                       const float* __restrict__ wp1,
                       unsigned int* wobs_bf, unsigned int* wg1_bf,
                       unsigned int* wp1_bf) {
    int i = blockIdx.x * blockDim.x + threadIdx.x;     // pair index
    if (i < Dd * Hh / 2) {
        const float2 v = *(const float2*)(wobs + (size_t)i * 2);
        wobs_bf[i] = pack2bf(v.x, v.y);
    }
    if (i < HQ * Hh / 2) {
        const float2 g = *(const float2*)(wg1 + (size_t)i * 2);
        const float2 p = *(const float2*)(wp1 + (size_t)i * 2);
        wg1_bf[i] = pack2bf(g.x, g.y);
        wp1_bf[i] = pack2bf(p.x, p.y);
    }
}

// Single pass over `hidden`: emit bf16 copy (A-side of the MLP GEMM) and
// hsum[t,h] = sum_b hidden[b,t,h] (mean folded into obs GEMM via linearity).
__global__ void k_hsum_pre(const float* __restrict__ hidden,
                           unsigned int* __restrict__ hsum_bf,
                           unsigned int* __restrict__ hidden_bf) {
    int i = blockIdx.x * blockDim.x + threadIdx.x;     // pair index over T*H/2
    if (i >= Tt * Hh / 2) return;
    float s0 = 0.f, s1 = 0.f;
#pragma unroll
    for (int b = 0; b < Bb; ++b) {
        const float2 v = *(const float2*)(hidden + (size_t)b * (Tt * Hh) + (size_t)i * 2);
        hidden_bf[(size_t)b * (Tt * Hh / 2) + i] = pack2bf(v.x, v.y);
        s0 += v.x; s1 += v.y;
    }
    hsum_bf[i] = pack2bf(s0, s1);
}
// Fallback (no bf16 copy of hidden).
__global__ void k_hsum(const float* __restrict__ hidden, unsigned int* __restrict__ hsum_bf) {
    int i = blockIdx.x * blockDim.x + threadIdx.x;
    if (i >= Tt * Hh / 2) return;
    float s0 = 0.f, s1 = 0.f;
#pragma unroll
    for (int b = 0; b < Bb; ++b) {
        const float2 v = *(const float2*)(hidden + (size_t)b * (Tt * Hh) + (size_t)i * 2);
        s0 += v.x; s1 += v.y;
    }
    hsum_bf[i] = pack2bf(s0, s1);
}

// Fused gate+prec MLP: one wave handles 32 rows (2 M-tiles) of [B*T].
// Layer-1 via WMMA over K=H in 8 N-groups of 128 cols (4 gate + 4 prec),
// layer-2 (dot with w2) fused in the epilogue, then sigmoid*softplus and
// atomic mean-accumulate into prec_sum[t].  PRE=true reads the bf16 copy of
// hidden (pure b128 gathers, no VALU conversion in the hot loop).
template <bool PRE>
__global__ __launch_bounds__(32) void k_mlp(const float* __restrict__ hidden,
        const unsigned short* __restrict__ hidden_bf,
        const unsigned short* __restrict__ wg1_bf,
        const unsigned short* __restrict__ wp1_bf,
        const float* __restrict__ b_g1, const float* __restrict__ w_g2,
        const float* __restrict__ b_g2,
        const float* __restrict__ b_p1, const float* __restrict__ w_p2,
        const float* __restrict__ b_p2,
        float* prec_sum) {
    const int lane = threadIdx.x;
    const int row0 = blockIdx.x * 32;
    const float* A0 = hidden + (size_t)row0 * Hh;
    const float* A1 = A0 + (size_t)16 * Hh;
    const unsigned short* Ab0 = hidden_bf + (size_t)row0 * Hh;
    const unsigned short* Ab1 = Ab0 + (size_t)16 * Hh;

    const floatx8 vzero = {0.f,0.f,0.f,0.f,0.f,0.f,0.f,0.f};
    float raccg[2][8], raccp[2][8];
#pragma unroll
    for (int mi = 0; mi < 2; ++mi)
#pragma unroll
        for (int r = 0; r < 8; ++r) { raccg[mi][r] = 0.f; raccp[mi][r] = 0.f; }

    for (int ng = 0; ng < 8; ++ng) {
        const unsigned short* W  = (ng < 4) ? wg1_bf : wp1_bf;
        const float*          w2 = (ng < 4) ? w_g2   : w_p2;
        const float*          b1 = (ng < 4) ? b_g1   : b_p1;
        const int col0 = (ng & 3) * 128;

        floatx8 acc0[8], acc1[8];
#pragma unroll
        for (int nt = 0; nt < 8; ++nt) { acc0[nt] = vzero; acc1[nt] = vzero; }

        for (int k = 0; k < Hh; k += 32) {
            bf16x16 a0, a1;
            if constexpr (PRE) {
                a0 = frag_bf16(Ab0 + k, Hh, lane);
                a1 = frag_bf16(Ab1 + k, Hh, lane);
            } else {
                a0 = frag_f32(A0 + k, Hh, lane);
                a1 = frag_f32(A1 + k, Hh, lane);
            }
#pragma unroll
            for (int nt = 0; nt < 8; ++nt) {
                bf16x16 b = frag_bf16(W + (size_t)(col0 + nt * 16) * Hh + k, Hh, lane);
                acc0[nt] = wmma_bf16(a0, b, acc0[nt]);
                acc1[nt] = wmma_bf16(a1, b, acc1[nt]);
            }
        }
        float (*racc)[8] = (ng < 4) ? raccg : raccp;
#pragma unroll
        for (int nt = 0; nt < 8; ++nt) {
            int col = col0 + nt * 16 + (lane & 15);
            float w2v = w2[col];
            float b1v = b1[col];
#pragma unroll
            for (int r = 0; r < 8; ++r) {
                racc[0][r] += fmaxf(acc0[nt][r] + b1v, 0.f) * w2v;
                racc[1][r] += fmaxf(acc1[nt][r] + b1v, 0.f) * w2v;
            }
        }
    }
    // reduce over the 16 lanes holding the same rows (columns of the tile)
#pragma unroll
    for (int mi = 0; mi < 2; ++mi)
#pragma unroll
        for (int r = 0; r < 8; ++r)
#pragma unroll
            for (int m = 1; m < 16; m <<= 1) {
                raccg[mi][r] += __shfl_xor(raccg[mi][r], m, 32);
                raccp[mi][r] += __shfl_xor(raccp[mi][r], m, 32);
            }
    if ((lane & 15) == 0) {
        float bg = b_g2[0], bp = b_p2[0];
#pragma unroll
        for (int mi = 0; mi < 2; ++mi)
#pragma unroll
            for (int r = 0; r < 8; ++r) {
                int grow = row0 + mi * 16 + ((lane >> 4) << 3) + r;  // flat b*T+t
                int t = grow & (Tt - 1);
                float g = 1.f / (1.f + __expf(-(raccg[mi][r] + bg)));
                float x = raccp[mi][r] + bp;
                float p = (x > 0.f) ? x + log1pf(__expf(-x)) : log1pf(__expf(x));
                atomicAdd(&prec_sum[t], g * p * (1.0f / Bb));
            }
    }
}

// obs_mean[T,D] = hsum_bf16 @ W_obs_bf16^T * (1/B)
__global__ __launch_bounds__(32) void k_obs(const unsigned short* __restrict__ hsum_bf,
        const unsigned short* __restrict__ wobs_bf, float* __restrict__ obs_mean) {
    const int lane = threadIdx.x;
    const int row0 = blockIdx.x * 16;
    const unsigned short* A = hsum_bf + (size_t)row0 * Hh;
    const floatx8 vzero = {0.f,0.f,0.f,0.f,0.f,0.f,0.f,0.f};
    for (int ng = 0; ng < 2; ++ng) {
        floatx8 acc[8];
#pragma unroll
        for (int nt = 0; nt < 8; ++nt) acc[nt] = vzero;
        for (int k = 0; k < Hh; k += 32) {
            bf16x16 a = frag_bf16(A + k, Hh, lane);
#pragma unroll
            for (int nt = 0; nt < 8; ++nt) {
                bf16x16 b = frag_bf16(wobs_bf + (size_t)(ng * 128 + nt * 16) * Hh + k,
                                      Hh, lane);
                acc[nt] = wmma_bf16(a, b, acc[nt]);
            }
        }
#pragma unroll
        for (int nt = 0; nt < 8; ++nt)
#pragma unroll
            for (int r = 0; r < 8; ++r) {
                int m = ((lane >> 4) << 3) + r;
                int col = ng * 128 + nt * 16 + (lane & 15);
                obs_mean[(size_t)(row0 + m) * Dd + col] = acc[nt][r] * (1.0f / Bb);
            }
    }
}

// Per-row t: normalize obs_mean, scale by prec_mean, emit obs_beliefs (out0),
// radius, meaningful (out3), and bf16 o_angles for the sims GEMM.
__global__ __launch_bounds__(256) void k_post(const float* __restrict__ obs_mean,
        const float* __restrict__ prec_sum, float* __restrict__ d_out,
        float* __restrict__ radii, unsigned short* __restrict__ oang_bf) {
    const int lane = threadIdx.x & 31;
    const int t = blockIdx.x * 8 + (threadIdx.x >> 5);
    const float* row = obs_mean + (size_t)t * Dd;
    float v[8]; float ss = 0.f;
#pragma unroll
    for (int i = 0; i < 8; ++i) { v[i] = row[lane * 8 + i]; ss += v[i] * v[i]; }
#pragma unroll
    for (int m = 1; m < 32; m <<= 1) ss += __shfl_xor(ss, m, 32);
    float nrm = sqrtf(ss);
    float inv = 1.f / fmaxf(nrm, 1e-8f);          // F.normalize semantics
    float prec = prec_sum[t];
    float rad = fabsf(prec) * (nrm * inv);        // ||obs_beliefs||
    float invr = 1.f / fmaxf(rad, 1e-8f);         // clip(.,eps) semantics
#pragma unroll
    for (int i = 0; i < 8; ++i) {
        float ob = v[i] * inv * prec;
        d_out[(size_t)t * Dd + lane * 8 + i] = ob;
        oang_bf[(size_t)t * Dd + lane * 8 + i] = f2bf(ob * invr);
    }
    if (lane == 0) {
        radii[t] = rad;
        d_out[(size_t)Tt * Dd + 2 * Tt + t] = (rad > 0.05f) ? 1.0f : 0.0f; // meaningful
    }
}

// Normalize beliefs rows -> bf16 b_angles.
__global__ __launch_bounds__(256) void k_bnorm(const float* __restrict__ beliefs,
                                               unsigned short* __restrict__ bang_bf) {
    const int lane = threadIdx.x & 31;
    const int n = blockIdx.x * 8 + (threadIdx.x >> 5);
    const float* row = beliefs + (size_t)n * Dd;
    float v[8]; float ss = 0.f;
#pragma unroll
    for (int i = 0; i < 8; ++i) { v[i] = row[lane * 8 + i]; ss += v[i] * v[i]; }
#pragma unroll
    for (int m = 1; m < 32; m <<= 1) ss += __shfl_xor(ss, m, 32);
    float inv = 1.f / fmaxf(sqrtf(ss), 1e-8f);    // clip(norm, EPS)
#pragma unroll
    for (int i = 0; i < 8; ++i)
        bang_bf[(size_t)n * Dd + lane * 8 + i] = f2bf(v[i] * inv);
}

// Similarity GEMM + masked argmax.  grid = (T/16, NSLICES); each wave keeps its
// 8 A-fragments (16x256) register-resident and sweeps its N slice.
constexpr int NSLICES = 8;
__global__ __launch_bounds__(32) void k_sims(const unsigned short* __restrict__ oang_bf,
        const unsigned short* __restrict__ bang_bf, const int* __restrict__ mask,
        unsigned long long* bestpack) {
    const int lane = threadIdx.x;
    const int row0 = blockIdx.x * 16;
    const int n0 = blockIdx.y * (Nn / NSLICES);
    const floatx8 vzero = {0.f,0.f,0.f,0.f,0.f,0.f,0.f,0.f};

    bf16x16 afr[8];
#pragma unroll
    for (int ks = 0; ks < 8; ++ks)
        afr[ks] = frag_bf16(oang_bf + (size_t)row0 * Dd + ks * 32, Dd, lane);

    float best[8]; int bidx[8];
#pragma unroll
    for (int r = 0; r < 8; ++r) { best[r] = -3.0e38f; bidx[r] = 0; }

    for (int nt = 0; nt < (Nn / NSLICES) / 16; ++nt) {
        int n = n0 + nt * 16;
        __builtin_prefetch(bang_bf + (size_t)(n + 16) * Dd, 0, 1); // global_prefetch_b8
        int colm = mask[n + (lane & 15)];
        floatx8 acc = vzero;
#pragma unroll
        for (int ks = 0; ks < 8; ++ks) {
            bf16x16 b = frag_bf16(bang_bf + (size_t)n * Dd + ks * 32, Dd, lane);
            acc = wmma_bf16(afr[ks], b, acc);
        }
        if (colm) {
            int nidx = n + (lane & 15);
#pragma unroll
            for (int r = 0; r < 8; ++r)
                if (acc[r] > best[r]) { best[r] = acc[r]; bidx[r] = nidx; }
        }
    }
    // combine across the 16 lanes (columns) in each half-wave; first-index ties
#pragma unroll
    for (int r = 0; r < 8; ++r)
#pragma unroll
        for (int m = 1; m < 16; m <<= 1) {
            float ob = __shfl_xor(best[r], m, 32);
            int   oi = __shfl_xor(bidx[r], m, 32);
            if (ob > best[r] || (ob == best[r] && oi < bidx[r])) { best[r] = ob; bidx[r] = oi; }
        }
    if ((lane & 15) == 0) {
#pragma unroll
        for (int r = 0; r < 8; ++r) {
            if (best[r] > -1.0e38f) {
                int t = row0 + ((lane >> 4) << 3) + r;
                unsigned int ub = __float_as_uint(best[r]);
                unsigned int key = (ub & 0x80000000u) ? ~ub : (ub | 0x80000000u);
                unsigned long long pk = ((unsigned long long)key << 32)
                                      | (unsigned int)(0x7FFFFFFF - bidx[r]);
                atomicMax(&bestpack[t], pk);
            }
        }
    }
}

__global__ void k_final(const unsigned long long* __restrict__ bestpack,
                        const float* __restrict__ radii, const int* __restrict__ anyflag,
                        float* __restrict__ d_out) {
    int t = blockIdx.x * blockDim.x + threadIdx.x;
    if (t >= Tt) return;
    float* sims_out = d_out + (size_t)Tt * Dd;
    float* slots    = sims_out + Tt;
    bool anyb = (*anyflag) != 0;
    unsigned long long pk = bestpack[t];
    unsigned int key = (unsigned int)(pk >> 32);
    unsigned int ub = (key & 0x80000000u) ? (key ^ 0x80000000u) : ~key;
    float best = __uint_as_float(ub);
    int idx = 0x7FFFFFFF - (int)(unsigned int)(pk & 0xFFFFFFFFu);
    float bs = anyb ? best : 0.0f;
    int bg = anyb ? idx : -1;
    bool meaningful = radii[t] > 0.05f;
    bool matched = meaningful && (bs > 0.5f) && anyb;
    sims_out[t] = matched ? bs : 0.0f;
    slots[t]    = matched ? (float)bg : -1.0f;
}

// ---------------- launch ----------------
extern "C" void kernel_launch(void* const* d_in, const int* in_sizes, int n_in,
                              void* d_out, int out_size, void* d_ws, size_t ws_size,
                              hipStream_t stream) {
    const float* hidden   = (const float*)d_in[0];
    const float* beliefs  = (const float*)d_in[1];
    const int*   amask    = (const int*)d_in[2];
    const float* W_obs    = (const float*)d_in[3];
    const float* W_g1     = (const float*)d_in[4];
    const float* b_g1     = (const float*)d_in[5];
    const float* W_g2     = (const float*)d_in[6];
    const float* b_g2     = (const float*)d_in[7];
    const float* W_p1     = (const float*)d_in[8];
    const float* b_p1     = (const float*)d_in[9];
    const float* W_p2     = (const float*)d_in[10];
    const float* b_p2     = (const float*)d_in[11];
    float* out = (float*)d_out;

    char* ws = (char*)d_ws;
    size_t off = 0;
    auto alloc = [&](size_t bytes) -> void* {
        void* p = ws + off;
        off += (bytes + 255) & ~(size_t)255;
        return p;
    };
    float*              prec_sum  = (float*)alloc((size_t)Tt * 4);
    unsigned long long* bestpack  = (unsigned long long*)alloc((size_t)Tt * 8);
    int*                anyflag   = (int*)alloc(256);
    float*              radii     = (float*)alloc((size_t)Tt * 4);
    float*              obs_mean  = (float*)alloc((size_t)Tt * Dd * 4);
    unsigned short*     hsum_bf   = (unsigned short*)alloc((size_t)Tt * Hh * 2);
    unsigned short*     wobs_bf   = (unsigned short*)alloc((size_t)Dd * Hh * 2);
    unsigned short*     wg1_bf    = (unsigned short*)alloc((size_t)HQ * Hh * 2);
    unsigned short*     wp1_bf    = (unsigned short*)alloc((size_t)HQ * Hh * 2);
    unsigned short*     bang_bf   = (unsigned short*)alloc((size_t)Nn * Dd * 2);
    unsigned short*     oang_bf   = (unsigned short*)alloc((size_t)Tt * Dd * 2);
    // Optional 64 MB bf16 copy of hidden (removes conversion from MLP hot loop).
    size_t hid_bytes = (size_t)Bb * Tt * Hh * 2;
    bool pre = (off + hid_bytes) <= ws_size;
    unsigned short* hidden_bf = (unsigned short*)alloc(hid_bytes);
    (void)in_sizes; (void)n_in; (void)out_size;

    k_init <<<(Tt + 255) / 256, 256, 0, stream>>>(prec_sum, bestpack, anyflag);
    k_any  <<<(Nn + 255) / 256, 256, 0, stream>>>(amask, anyflag);
    k_cvtw <<<(HQ * Hh / 2 + 255) / 256, 256, 0, stream>>>(W_obs, W_g1, W_p1,
            (unsigned int*)wobs_bf, (unsigned int*)wg1_bf, (unsigned int*)wp1_bf);
    if (pre) {
        k_hsum_pre<<<(Tt * Hh / 2 + 255) / 256, 256, 0, stream>>>(
            hidden, (unsigned int*)hsum_bf, (unsigned int*)hidden_bf);
        k_mlp<true><<<(Bb * Tt) / 32, 32, 0, stream>>>(hidden, hidden_bf,
            wg1_bf, wp1_bf, b_g1, W_g2, b_g2, b_p1, W_p2, b_p2, prec_sum);
    } else {
        k_hsum<<<(Tt * Hh / 2 + 255) / 256, 256, 0, stream>>>(
            hidden, (unsigned int*)hsum_bf);
        k_mlp<false><<<(Bb * Tt) / 32, 32, 0, stream>>>(hidden, hidden_bf,
            wg1_bf, wp1_bf, b_g1, W_g2, b_g2, b_p1, W_p2, b_p2, prec_sum);
    }
    k_obs  <<<Tt / 16, 32, 0, stream>>>(hsum_bf, wobs_bf, obs_mean);
    k_post <<<Tt / 8, 256, 0, stream>>>(obs_mean, prec_sum, out, radii, oang_bf);
    k_bnorm<<<Nn / 8, 256, 0, stream>>>(beliefs, bang_bf);
    k_sims <<<dim3(Tt / 16, NSLICES), 32, 0, stream>>>(oang_bf, bang_bf, amask, bestpack);
    k_final<<<(Tt + 255) / 256, 256, 0, stream>>>(bestpack, radii, anyflag, out);
}
